// ModelF_20332375179872
// MI455X (gfx1250) — matrix operand order repaired and checked by reading, not compiled, not run on hardware
//
#include <hip/hip_runtime.h>

// ---------- types ----------
typedef __bf16 bf16;
typedef bf16 bf16x4  __attribute__((ext_vector_type(4)));
typedef bf16 bf16x8  __attribute__((ext_vector_type(8)));
typedef bf16 bf16x16 __attribute__((ext_vector_type(16)));
typedef float f32x4  __attribute__((ext_vector_type(4)));
typedef float f32x8  __attribute__((ext_vector_type(8)));

#define BATCH 512
#define NPATCH 196
#define LTOK 256
#define EDIM 768
#define PDIM 512
#define KSEL 98
#define BROWS 128                      // rows per workgroup in projection GEMM
#define NEG_INF_F (-3.402823466e38f)

union FragBF { bf16x16 v; bf16x8 h[2]; };

static __device__ __forceinline__ f32x8 wmma_bf16(const FragBF& a, const FragBF& b, f32x8 c) {
  // D = A(16x32 bf16) x B(32x16 bf16) + C(16x16 f32)
  return __builtin_amdgcn_wmma_f32_16x16x32_bf16(false, a.v, false, b.v, (short)0, c, false, false);
}

// gfx1250 async copy: 16B global -> LDS, tracked on ASYNCcnt (no VGPR round-trip).
// VDST vgpr holds LDS byte offset (generic LDS pointer low 32 bits == LDS offset).
static __device__ __forceinline__ void async_copy_b128(bf16* lds_dst, const bf16* gsrc) {
  unsigned loff = (unsigned)(unsigned long long)lds_dst;
  asm volatile("global_load_async_to_lds_b128 %0, %1, off"
               :: "v"(loff), "v"(gsrc)
               : "memory");
}
static __device__ __forceinline__ void wait_async0() {
  asm volatile("s_wait_asynccnt 0x0" ::: "memory");
}

// ---------- kernel 1: W (E,P) f32 -> Wt (P,E) bf16, for both weights ----------
__global__ void __launch_bounds__(256) prep_w(const float* __restrict__ Wpp,
                                              const float* __restrict__ Wtp,
                                              bf16* __restrict__ WtA,
                                              bf16* __restrict__ WtB) {
  int idx = blockIdx.x * 256 + threadIdx.x;          // 0 .. 2*E*P-1
  const int total1 = EDIM * PDIM;
  const float* W = (idx < total1) ? Wpp : Wtp;
  bf16* O        = (idx < total1) ? WtA : WtB;
  int r = (idx < total1) ? idx : idx - total1;
  int p = r / EDIM;
  int e = r - p * EDIM;
  O[p * EDIM + e] = (bf16)W[e * PDIM + p];
}

// ---------- kernel 2: C = l2norm(A[M,768] @ W[768,512] + b) -> bf16 [M,512] ----------
// One workgroup: 128 rows x 512 cols, 1024 threads (32 waves).
// Wave w: row tile (w>>2)*16, col group (w&3)*128 (8 WMMA C-tiles, 64 acc VGPRs).
__global__ void __launch_bounds__(1024) proj_norm_gemm(const float* __restrict__ A,
                                                       const bf16*  __restrict__ Wt,   // [512][768]
                                                       const float* __restrict__ bias, // [512]
                                                       bf16* __restrict__ O) {         // [M][512]
  __shared__ bf16 As[BROWS][32];
  __shared__ bf16 Bs[512][32];
  __shared__ float rowSS[BROWS];

  const int tid  = threadIdx.x;
  const int w    = tid >> 5;          // 0..31
  const int lane = tid & 31;
  const int half = lane >> 4;
  const int lrow = lane & 15;
  const int r0   = (w >> 2) * 16;     // 0..112
  const int cg   = (w & 3) * 128;
  const long rowBase = (long)blockIdx.x * BROWS;

  f32x8 acc[8];
#pragma unroll
  for (int t = 0; t < 8; ++t)
#pragma unroll
    for (int i = 0; i < 8; ++i) acc[t][i] = 0.0f;

  for (int kk = 0; kk < EDIM; kk += 32) {
    // stage B tile via async global->LDS (512 cols x 32 k bf16, k-contiguous per col)
    if (tid < 512) {
      const bf16* src = &Wt[tid * EDIM + kk];
      bf16* dst = &Bs[tid][0];
#pragma unroll
      for (int i = 0; i < 4; ++i) async_copy_b128(dst + i * 8, src + i * 8);
    }
    // stage A tile (128x32 f32 -> bf16) through VGPRs (needs conversion)
    {
      int ar = tid >> 3;               // 0..127
      int ac = (tid & 7) * 4;          // 0..28
      const float* ap = &A[(rowBase + ar) * EDIM + kk + ac];
      if (kk + 32 < EDIM) __builtin_prefetch(ap + 32, 0, 1);
      f32x4 v = *(const f32x4*)ap;
      bf16x4 hv;
      hv[0] = (bf16)v[0]; hv[1] = (bf16)v[1]; hv[2] = (bf16)v[2]; hv[3] = (bf16)v[3];
      *(bf16x4*)&As[ar][ac] = hv;
    }
    wait_async0();
    __syncthreads();

    FragBF af;
    af.h[0] = *(const bf16x8*)&As[r0 + lrow][half * 8];
    af.h[1] = *(const bf16x8*)&As[r0 + lrow][16 + half * 8];
#pragma unroll
    for (int t = 0; t < 8; ++t) {
      FragBF bf;
      const bf16* bp = &Bs[cg + t * 16 + lrow][half * 8];
      bf.h[0] = *(const bf16x8*)bp;
      bf.h[1] = *(const bf16x8*)(bp + 16);
      acc[t] = wmma_bf16(af, bf, acc[t]);
    }
    __syncthreads();
  }

  // bias + sum-of-squares per row
  float ssq[8];
#pragma unroll
  for (int r = 0; r < 8; ++r) ssq[r] = 0.0f;
#pragma unroll
  for (int t = 0; t < 8; ++t) {
    float bb = bias[cg + t * 16 + lrow];
#pragma unroll
    for (int r = 0; r < 8; ++r) {
      float v = acc[t][r] + bb;
      acc[t][r] = v;
      ssq[r] += v * v;
    }
  }
  if (tid < BROWS) rowSS[tid] = 0.0f;
  __syncthreads();
#pragma unroll
  for (int r = 0; r < 8; ++r) atomicAdd(&rowSS[r0 + r + half * 8], ssq[r]);
  __syncthreads();

#pragma unroll
  for (int r = 0; r < 8; ++r) {
    int rowl = r0 + r + half * 8;
    float s = sqrtf(rowSS[rowl]);
    float inv = 1.0f / fmaxf(s, 1e-12f);
    long row = rowBase + rowl;
#pragma unroll
    for (int t = 0; t < 8; ++t)
      O[row * PDIM + cg + t * 16 + lrow] = (bf16)(acc[t][r] * inv);
  }
}

// ---------- kernel 3: sim = pf @ tf^T, mask, max over tokens -> scores ----------
// grid = B * 13 row tiles. One workgroup: 16 patch rows x 256 token cols.
__global__ void __launch_bounds__(256) sim_max(const bf16* __restrict__ pf,   // [B*196][512]
                                               const bf16* __restrict__ tf,   // [B*256][512]
                                               const int*  __restrict__ amask,// [B][256]
                                               float* __restrict__ scores) {  // [B][196]
  __shared__ bf16 As[16][32];
  __shared__ bf16 Bs[256][32];
  __shared__ float red[8][16];

  const int b  = blockIdx.x / 13;
  const int rt = blockIdx.x % 13;
  const int tid  = threadIdx.x;
  const int w    = tid >> 5;
  const int lane = tid & 31;
  const int half = lane >> 4;
  const int lrow = lane & 15;

  f32x8 acc[2];
#pragma unroll
  for (int t = 0; t < 2; ++t)
#pragma unroll
    for (int i = 0; i < 8; ++i) acc[t][i] = 0.0f;

  for (int kk = 0; kk < PDIM; kk += 32) {
    // stage tf tile via async global->LDS: one token row (32 k bf16) per thread
    {
      const bf16* src = &tf[((long)b * LTOK + tid) * PDIM + kk];
      bf16* dst = &Bs[tid][0];
#pragma unroll
      for (int i = 0; i < 4; ++i) async_copy_b128(dst + i * 8, src + i * 8);
    }
    // stage pf tile (bounds-checked, zero-fill past row 195)
    {
      int i  = tid >> 4;          // 0..15
      int kc = (tid & 15) * 2;
      int n  = rt * 16 + i;
      bf16 v0 = (bf16)0.0f, v1 = (bf16)0.0f;
      if (n < NPATCH) {
        const bf16* src = &pf[((long)b * NPATCH + n) * PDIM + kk + kc];
        v0 = src[0]; v1 = src[1];
      }
      As[i][kc] = v0; As[i][kc + 1] = v1;
    }
    wait_async0();
    __syncthreads();

    FragBF af;
    af.h[0] = *(const bf16x8*)&As[lrow][half * 8];
    af.h[1] = *(const bf16x8*)&As[lrow][16 + half * 8];
#pragma unroll
    for (int t = 0; t < 2; ++t) {
      FragBF bf;
      const bf16* bp = &Bs[(2 * w + t) * 16 + lrow][half * 8];
      bf.h[0] = *(const bf16x8*)bp;
      bf.h[1] = *(const bf16x8*)(bp + 16);
      acc[t] = wmma_bf16(af, bf, acc[t]);
    }
    __syncthreads();
  }

  // pad-mask + row max
  int mk0 = amask[b * LTOK + (2 * w + 0) * 16 + lrow];
  int mk1 = amask[b * LTOK + (2 * w + 1) * 16 + lrow];
#pragma unroll
  for (int r = 0; r < 8; ++r) {
    float v0 = (mk0 != 0) ? acc[0][r] : NEG_INF_F;
    float v1 = (mk1 != 0) ? acc[1][r] : NEG_INF_F;
    float v = fmaxf(v0, v1);
    v = fmaxf(v, __shfl_xor(v, 8, 32));
    v = fmaxf(v, __shfl_xor(v, 4, 32));
    v = fmaxf(v, __shfl_xor(v, 2, 32));
    v = fmaxf(v, __shfl_xor(v, 1, 32));
    if (lrow == 0) red[w][r + half * 8] = v;
  }
  __syncthreads();
  if (tid < 16) {
    float m = red[0][tid];
#pragma unroll
    for (int ww = 1; ww < 8; ++ww) m = fmaxf(m, red[ww][tid]);
    int n = rt * 16 + tid;
    if (n < NPATCH) scores[b * NPATCH + n] = m;
  }
}

// ---------- kernel 4: top-98 (jax tie-break: lower index first), sort idx, gather ----------
__global__ void __launch_bounds__(256) topk_gather(const float* __restrict__ scores, // [B][196]
                                                   const float* __restrict__ state,  // [B][197][768]
                                                   float* __restrict__ out) {        // [B][99][768]
  __shared__ float s[NPATCH];
  __shared__ int flag[NPATCH];
  __shared__ int list[KSEL];
  const int b = blockIdx.x;
  const int tid = threadIdx.x;

  if (tid < NPATCH) s[tid] = scores[b * NPATCH + tid];
  __syncthreads();
  if (tid < NPATCH) {
    float si = s[tid];
    int rank = 0;
    for (int j = 0; j < NPATCH; ++j) {
      float sj = s[j];
      rank += (sj > si) || ((sj == si) && (j < tid));
    }
    flag[tid] = (rank < KSEL) ? 1 : 0;
  }
  __syncthreads();
  if (tid < NPATCH && flag[tid]) {
    int pos = 0;
    for (int j = 0; j < tid; ++j) pos += flag[j];
    list[pos] = tid;   // ascending by construction
  }
  __syncthreads();

  // gather: row 0 = CLS, rows 1..98 = selected patches (state rows 1+idx)
  const f32x4* src = (const f32x4*)state;
  f32x4* dst = (f32x4*)out;
  const int ROWV = EDIM / 4;                   // 192 float4 per row
  for (int idx = tid; idx < (KSEL + 1) * ROWV; idx += 256) {
    int r = idx / ROWV;
    int c = idx - r * ROWV;
    int srcRow = (r == 0) ? 0 : (1 + list[r - 1]);
    dst[((long)b * (KSEL + 1) + r) * ROWV + c] =
        src[((long)b * (NPATCH + 1) + srcRow) * ROWV + c];
  }
}

// ---------- host ----------
extern "C" void kernel_launch(void* const* d_in, const int* in_sizes, int n_in,
                              void* d_out, int out_size, void* d_ws, size_t ws_size,
                              hipStream_t stream) {
  const float* patch = (const float*)d_in[0];   // (B, N, E)
  const float* state = (const float*)d_in[1];   // (B, N+1, E)
  const float* token = (const float*)d_in[2];   // (B, L, E)
  const float* Wpp   = (const float*)d_in[3];   // (E, P)
  const float* bpp   = (const float*)d_in[4];   // (P,)
  const float* Wtp   = (const float*)d_in[5];   // (E, P)
  const float* btp   = (const float*)d_in[6];   // (P,)
  const int*   amask = (const int*)d_in[7];     // (B, L)
  float* out = (float*)d_out;                   // (B, K+1, E) f32

  char* ws = (char*)d_ws;
  size_t off = 0;
  bf16* WtA = (bf16*)(ws + off); off += (size_t)PDIM * EDIM * 2;
  bf16* WtB = (bf16*)(ws + off); off += (size_t)PDIM * EDIM * 2;
  bf16* pf  = (bf16*)(ws + off); off += (size_t)BATCH * NPATCH * PDIM * 2;
  bf16* tfb = (bf16*)(ws + off); off += (size_t)BATCH * LTOK * PDIM * 2;
  float* sc = (float*)(ws + off); off += (size_t)BATCH * NPATCH * 4;
  (void)ws_size; (void)in_sizes; (void)n_in; (void)out_size;

  prep_w<<<(2 * EDIM * PDIM) / 256, 256, 0, stream>>>(Wpp, Wtp, WtA, WtB);
  proj_norm_gemm<<<(BATCH * NPATCH) / BROWS, 1024, 0, stream>>>(patch, WtA, bpp, pf);
  proj_norm_gemm<<<(BATCH * LTOK) / BROWS, 1024, 0, stream>>>(token, WtB, btp, tfb);
  sim_max<<<BATCH * 13, 256, 0, stream>>>(pf, tfb, amask, sc);
  topk_gather<<<BATCH, 256, 0, stream>>>(sc, state, out);
}
